// Attention_11003706212887
// MI455X (gfx1250) — matrix operand order repaired and checked by reading, not compiled, not run on hardware
//
#include <hip/hip_runtime.h>

// ---------------------------------------------------------------------------
// Problem constants (match reference): B=4, S=2048, D=1024
// ---------------------------------------------------------------------------
constexpr int BB = 4;
constexpr int SS = 2048;
constexpr int DD = 1024;

typedef __attribute__((ext_vector_type(16))) __bf16 v16bf;
typedef __attribute__((ext_vector_type(8)))  __bf16 v8bf;
typedef __attribute__((ext_vector_type(8)))  float  v8f;
typedef int v4i __attribute__((vector_size(16)));   // matches async-LDS builtin param

#if __has_builtin(__builtin_amdgcn_global_load_async_to_lds_b128)
#define ASYNC_LDS 1
#endif

// ---------------------------------------------------------------------------
// Fragment layouts this code uses for V_WMMA_F32_16X16X32_BF16 (contiguous k, not the ISA table order):
// A (16x32 MxK): lane L<16: row M=L, elems 0..15=K+0..15;
//                lane L+16: same row, K+16..31.
// B (32x16 KxN): lane c=L&15 -> col N=c, elems 0..15 = K + 16*(L>>4) .. +15
//                (16 contiguous in a K-major row => single 32B load).
// C/D: VGPR r: lanes 0-15 -> M=r, lanes 16-31 -> M=r+8; N = L&15.
// ---------------------------------------------------------------------------
__device__ __forceinline__ v16bf load_a_lds(const __bf16* rowPtr, int kBase, int lane) {
    const int off = (lane < 16) ? 0 : 16;
    v16bf a;
#pragma unroll
    for (int i = 0; i < 16; ++i) {
        a[i]     = rowPtr[kBase + off + i];
    }
    return a;
}

__device__ __forceinline__ v8f wmma_bf16(v16bf a, v16bf b, v8f c) {
    return __builtin_amdgcn_wmma_f32_16x16x32_bf16(false, a, false, b, (short)0, c,
                                                   false, false);
}

// ---------------------------------------------------------------------------
// Kernel 1: weight transpose + bf16 convert.  WT[j][i] = bf16(W[i][j])
// ---------------------------------------------------------------------------
__global__ void wt_kernel(const float* __restrict__ W, __bf16* __restrict__ WT) {
    int idx = blockIdx.x * 256 + threadIdx.x;
    int i = idx >> 10;                       // DD == 1024
    int j = idx & (DD - 1);
    WT[(size_t)j * DD + i] = (__bf16)W[(size_t)i * DD + j];
}

// ---------------------------------------------------------------------------
// Kernel 2: projection  out = X @ W + bias.
// Block = 8 waves sharing one 16-row X tile staged in LDS (fused f32->bf16);
// wave w owns a 64-column slice -> block computes a 16x512 output tile.
// vtFlag=1 writes V transposed per batch: VT[b][d][s].
// ---------------------------------------------------------------------------
__global__ void project_kernel(const float* __restrict__ X,
                               const __bf16* __restrict__ WT,
                               const float* __restrict__ bias,
                               __bf16* __restrict__ out, int vtFlag) {
    __shared__ __bf16 Xs[16][DD];                  // 32 KB

    const int lane = threadIdx.x & 31;
    const int wave = threadIdx.x >> 5;             // 0..7
    const int rowBase = blockIdx.x * 16;           // row tile within batch
    const int colTile = blockIdx.y * 512 + wave * 64;
    const int b = blockIdx.z;
    const int colL = lane & 15, sel = lane >> 4;

    // ---- cooperative load + convert: 16x1024 f32 -> bf16 in LDS (contiguous)
    {
        const float* gsrc = X + ((size_t)b * SS + rowBase) * (size_t)DD;
        __bf16* lbase = &Xs[0][0];
#pragma unroll
        for (int i = 0; i < 8; ++i) {
            const int off = threadIdx.x * 64 + i * 8;
            v8f f = *(const v8f*)(gsrc + off);
            v8bf h;
#pragma unroll
            for (int j = 0; j < 8; ++j) h[j] = (__bf16)f[j];
            *(v8bf*)(lbase + off) = h;
        }
    }
    __syncthreads();

    // ---- B-operand row pointers (weights, K-contiguous rows; L2-resident)
    const __bf16* wrow[4];
#pragma unroll
    for (int t = 0; t < 4; ++t)
        wrow[t] = WT + (size_t)(colTile + 16 * t + colL) * DD + 16 * sel;

    v8f acc[4] = {};
    v16bf bw[4];
#pragma unroll
    for (int t = 0; t < 4; ++t) bw[t] = *(const v16bf*)(wrow[t]);

    for (int d0 = 0; d0 < DD; d0 += 32) {
        v16bf a = load_a_lds(&Xs[colL][0], d0, lane);
        v16bf bn[4];
        const bool more = (d0 + 32) < DD;
        if (more) {
#pragma unroll
            for (int t = 0; t < 4; ++t) bn[t] = *(const v16bf*)(wrow[t] + d0 + 32);
        }
#pragma unroll
        for (int t = 0; t < 4; ++t) acc[t] = wmma_bf16(a, bw[t], acc[t]);
        if (more) {
#pragma unroll
            for (int t = 0; t < 4; ++t) bw[t] = bn[t];
        }
    }

#pragma unroll
    for (int t = 0; t < 4; ++t) {
        const int col = colTile + 16 * t + colL;
        const float bv = bias[col];
#pragma unroll
        for (int r = 0; r < 8; ++r) {
            const int row = r + 8 * sel;
            const float v = acc[t][r] + bv;
            if (!vtFlag)
                out[((size_t)b * SS + rowBase + row) * DD + col] = (__bf16)v;
            else
                out[((size_t)b * DD + col) * SS + rowBase + row] = (__bf16)v;
        }
    }
}

// ---------------------------------------------------------------------------
// Kernel 3: causal flash attention.
// Block = 8 waves, one 16-query tile; Q tile staged once in LDS (async path).
// Per 128-key block: wave w -> one 16x16 score tile; wave 0 -> online softmax
// (32 lanes, shfl combine); every wave -> P@V on its 128-column slice.
// ---------------------------------------------------------------------------
__global__ void flash_kernel(const __bf16* __restrict__ Qp,
                             const __bf16* __restrict__ Kp,
                             const __bf16* __restrict__ VT,
                             float* __restrict__ out) {
    __shared__ __bf16 Qs[16][DD];                  // 32 KB
    __shared__ float  sc[16][128];                 // 8 KB
    __shared__ __bf16 Pl[16][128];                 // 4 KB
    __shared__ float  m_s[16], l_s[16], alpha_s[16];

    const int lane = threadIdx.x & 31;
    const int wave = threadIdx.x >> 5;             // 0..7
    const int qBase = blockIdx.x * 16;
    const int batch = blockIdx.y;
    const int colL = lane & 15, sel = lane >> 4;
    const int dBase = wave * 128;                  // this wave's output columns

    const __bf16* Kb = Kp + (size_t)batch * SS * DD;
    const __bf16* Vb = VT + (size_t)batch * DD * (size_t)SS;

    // ---- stage Q tile (16x1024 bf16, one contiguous 32KB region) into LDS
    {
        const __bf16* gsrc = Qp + ((size_t)batch * SS + qBase) * (size_t)DD;
        __bf16* lbase = &Qs[0][0];
#ifdef ASYNC_LDS
#pragma unroll
        for (int i = 0; i < 8; ++i) {
            const int off = threadIdx.x * 64 + i * 8;   // 16B chunks
            __builtin_amdgcn_global_load_async_to_lds_b128(
                (v4i*)(gsrc + off), (v4i*)(lbase + off), 0, 0);
        }
        asm volatile("s_wait_asynccnt 0x0" ::: "memory");
#else
#pragma unroll
        for (int i = 0; i < 8; ++i) {
            const int off = threadIdx.x * 64 + i * 8;
            *(v8bf*)(lbase + off) = *(const v8bf*)(gsrc + off);
        }
#endif
    }
    if (threadIdx.x < 16) { m_s[threadIdx.x] = -3.0e38f; l_s[threadIdx.x] = 0.f; }
    __syncthreads();

    v8f acc[8] = {};

    const int nkb = (qBase + 15) / 128 + 1;        // causal: skip masked blocks
    for (int kb = 0; kb < nkb; ++kb) {
        const int kBase = kb * 128;

        // ---- warm L2 for the next key block while we compute this one
        if (kb + 1 < nkb) {
            const __bf16* nk = Kb + (size_t)(kBase + 128 + 16 * wave + colL) * DD
                                  + (size_t)sel * 512;
#pragma unroll
            for (int i = 0; i < 4; ++i)
                __builtin_prefetch(nk + i * 128, 0, 1);
#pragma unroll
            for (int i = 0; i < 4; ++i)
                __builtin_prefetch(Vb + (size_t)(dBase + 4 * lane + i) * SS
                                      + (kBase + 128), 0, 1);
        }

        // ---- scores: wave w computes Q_tile . K[kBase+16w..+15]^T (pipelined)
        v8f accS = {};
        const __bf16* brow = Kb + (size_t)(kBase + 16 * wave + colL) * DD + 16 * sel;
        v16bf bk = *(const v16bf*)(brow);
        for (int d0 = 0; d0 < DD; d0 += 32) {
            v16bf a = load_a_lds(&Qs[colL][0], d0, lane);
            v16bf bn = bk;
            if (d0 + 32 < DD) bn = *(const v16bf*)(brow + d0 + 32);
            accS = wmma_bf16(a, bk, accS);
            bk = bn;
        }
        const int key = kBase + 16 * wave + colL;
#pragma unroll
        for (int r = 0; r < 8; ++r) {
            const int row = r + 8 * sel;
            float v = accS[r] * 0.03125f;          // 1/sqrt(1024)
            if (key > qBase + row) v -= 1e9f;      // causal mask (matches ref)
            sc[row][16 * wave + colL] = v;
        }
        __syncthreads();

        // ---- online softmax: wave 0, 2 lanes per row, shfl_xor(16) combine
        if (threadIdx.x < 32) {
            const int row = threadIdx.x & 15;
            const int h = threadIdx.x >> 4;        // column half [64h, 64h+64)
            const float mo = m_s[row];
            float m = mo;
#pragma unroll 8
            for (int j = 0; j < 64; ++j) m = fmaxf(m, sc[row][64 * h + j]);
            m = fmaxf(m, __shfl_xor(m, 16, 32));
            float sum = 0.f;
#pragma unroll 8
            for (int j = 0; j < 64; ++j) {
                const float p = __expf(sc[row][64 * h + j] - m);
                Pl[row][64 * h + j] = (__bf16)p;
                sum += p;
            }
            sum += __shfl_xor(sum, 16, 32);
            if (h == 0) {
                const float al = __expf(mo - m);
                m_s[row] = m;
                alpha_s[row] = al;
                l_s[row] = l_s[row] * al + sum;
            }
        }
        __syncthreads();

        // ---- rescale accumulators, then P @ V on this wave's column slice
        float al[8];
#pragma unroll
        for (int r = 0; r < 8; ++r) al[r] = alpha_s[r + 8 * sel];
#pragma unroll
        for (int t = 0; t < 8; ++t)
#pragma unroll
            for (int r = 0; r < 8; ++r) acc[t][r] *= al[r];

        for (int ks = 0; ks < 4; ++ks) {
            const int offA = (lane < 16) ? 0 : 16;
            v16bf pa;
#pragma unroll
            for (int i = 0; i < 16; ++i) {
                pa[i]     = Pl[colL][32 * ks + offA + i];
            }
#pragma unroll
            for (int t = 0; t < 8; ++t) {
                const __bf16* vp = Vb + (size_t)(dBase + 16 * t + colL) * SS
                                      + kBase + 32 * ks + 16 * sel;
                v16bf bv = *(const v16bf*)(vp);
                acc[t] = wmma_bf16(pa, bv, acc[t]);
            }
        }
        __syncthreads();                           // protect sc/Pl next block
    }

    // ---- finalize: out = acc / l
    float ll[8];
#pragma unroll
    for (int r = 0; r < 8; ++r) ll[r] = l_s[r + 8 * sel];
#pragma unroll
    for (int t = 0; t < 8; ++t) {
        const int dcol = dBase + 16 * t + colL;
        float* op = out + ((size_t)batch * SS + qBase) * DD + dcol;
#pragma unroll
        for (int r = 0; r < 8; ++r) {
            const int row = r + 8 * sel;
            op[(size_t)row * DD] = acc[t][r] / ll[r];
        }
    }
}

// ---------------------------------------------------------------------------
// Host launcher
// in: 0=q 1=k 2=v 3=mask(unused; causal applied analytically) 4=Wq 5=bq
//     6=Wk 7=bk 8=Wv 9=bv        out: [B,S,D] fp32
// ---------------------------------------------------------------------------
extern "C" void kernel_launch(void* const* d_in, const int* in_sizes, int n_in,
                              void* d_out, int out_size, void* d_ws, size_t ws_size,
                              hipStream_t stream) {
    const float* q  = (const float*)d_in[0];
    const float* k  = (const float*)d_in[1];
    const float* v  = (const float*)d_in[2];
    const float* Wq = (const float*)d_in[4];
    const float* bq = (const float*)d_in[5];
    const float* Wk = (const float*)d_in[6];
    const float* bk = (const float*)d_in[7];
    const float* Wv = (const float*)d_in[8];
    const float* bv = (const float*)d_in[9];

    char* ws = (char*)d_ws;
    const size_t wBytes = (size_t)DD * DD * sizeof(__bf16);      // 2 MB each
    const size_t pBytes = (size_t)BB * SS * DD * sizeof(__bf16); // 16 MB each
    __bf16* WqT = (__bf16*)ws;              ws += wBytes;
    __bf16* WkT = (__bf16*)ws;              ws += wBytes;
    __bf16* WvT = (__bf16*)ws;              ws += wBytes;
    __bf16* Qp  = (__bf16*)ws;              ws += pBytes;
    __bf16* Kp  = (__bf16*)ws;              ws += pBytes;
    __bf16* VT  = (__bf16*)ws;              ws += pBytes;

    const int tBlocks = (DD * DD) / 256;
    wt_kernel<<<tBlocks, 256, 0, stream>>>(Wq, WqT);
    wt_kernel<<<tBlocks, 256, 0, stream>>>(Wk, WkT);
    wt_kernel<<<tBlocks, 256, 0, stream>>>(Wv, WvT);

    dim3 gp(SS / 16, DD / 512, BB);
    project_kernel<<<gp, 256, 0, stream>>>(q, WqT, bq, Qp, 0);
    project_kernel<<<gp, 256, 0, stream>>>(k, WkT, bk, Kp, 0);
    project_kernel<<<gp, 256, 0, stream>>>(v, WvT, bv, VT, 1);

    dim3 gf(SS / 16, BB);
    flash_kernel<<<gf, 256, 0, stream>>>(Qp, Kp, VT, (float*)d_out);
}
